// MoEMLP_83554293776402
// MI455X (gfx1250) — matrix operand order repaired and checked by reading, not compiled, not run on hardware
//
#include <hip/hip_runtime.h>
#include <hip/hip_bf16.h>
#include <stdint.h>

// Problem constants (match reference): E=8, K=2, D=2048, H=5632, S=4096
#define S_TOK 4096
#define DDIM  2048
#define HDIM  5632
#define NE    8
#define TOPK  2

typedef __attribute__((ext_vector_type(16))) __bf16 v16bf;
typedef __attribute__((ext_vector_type(8)))  __bf16 bf8_t;
typedef __attribute__((ext_vector_type(8)))  float  v8f;

// GEMM tiling
#define BM 64
#define BN 128
#define BK 32
#define KST 40              // padded LDS row stride (elements): 80B, 16B-aligned, conflict-free
#define XBUFB (BM * KST * 2)    // bytes per X LDS buffer  = 5120
#define WBUFB (BN * KST * 2)    // bytes per W LDS buffer  = 10240

// ---------------- CDNA5 async copy + wait helpers (inline asm) ----------------
__device__ __forceinline__ void async_b128(uint32_t lds_addr, uint64_t gaddr) {
  asm volatile("global_load_async_to_lds_b128 %0, %1, off"
               :: "v"(lds_addr), "v"(gaddr) : "memory");
}
template <int N>
__device__ __forceinline__ void wait_async() {
  asm volatile("s_wait_asynccnt %0" :: "n"(N) : "memory");
}

// ---------------- WMMA helper ----------------
__device__ __forceinline__ v8f wmma_bf16(v16bf a, v16bf b, v8f c) {
  return __builtin_amdgcn_wmma_f32_16x16x32_bf16(false, a, false, b, (short)0, c,
                                                 false, false);
}

union FragU { v16bf v; bf8_t h[2]; };

// A fragment (16x32, M x K): lane L -> row M=L&15; K chunks {half*8 .. +7} and {16+half*8 ..}
__device__ __forceinline__ v16bf afrag(const __bf16* p, int lane) {
  const __bf16* q = p + (lane & 15) * KST + ((lane >> 4) << 3);
  FragU f;
  f.h[0] = *(const bf8_t*)q;
  f.h[1] = *(const bf8_t*)(q + 16);
  return f.v;
}
// B fragment (32x16, K x N): lane L -> col N=L&15; K chunks {half*8 .. +7} and {16+half*8 ..}
__device__ __forceinline__ v16bf bfrag(const __bf16* p, int lane) {
  const __bf16* q = p + (lane & 15) * KST + ((lane >> 4) << 3);
  FragU f;
  f.h[0] = *(const bf8_t*)q;
  f.h[1] = *(const bf8_t*)(q + 16);
  return f.v;
}

// ---------------- small utility kernels ----------------
__global__ void zero_k(uint4* p, long long n4) {
  uint4 z; z.x = z.y = z.z = z.w = 0u;
  for (long long i = (long long)blockIdx.x * blockDim.x + threadIdx.x; i < n4;
       i += (long long)gridDim.x * blockDim.x)
    p[i] = z;
}

__global__ void finalize_k(const float* __restrict__ f, __bf16* __restrict__ o,
                           long long n) {
  for (long long i = (long long)blockIdx.x * blockDim.x + threadIdx.x; i < n;
       i += (long long)gridDim.x * blockDim.x)
    o[i] = (__bf16)f[i];
}

// ---------------- router: logits -> softmax -> top2 ----------------
__global__ __launch_bounds__(256) void router_k(const __bf16* __restrict__ x,
                                                const __bf16* __restrict__ gk,
                                                int* __restrict__ eid,
                                                float* __restrict__ prob,
                                                int* __restrict__ cnt) {
  const int lane = threadIdx.x & 31;
  const int s = blockIdx.x * 8 + (threadIdx.x >> 5);   // one wave32 per token
  const __bf16* xr = x + (size_t)s * DDIM;
  float acc[NE];
#pragma unroll
  for (int e = 0; e < NE; ++e) acc[e] = 0.f;
  for (int d = lane; d < DDIM; d += 32) {
    float xv = (float)xr[d];
    bf8_t g = *(const bf8_t*)(gk + (size_t)d * NE);    // 16B row of gate_kernel
#pragma unroll
    for (int e = 0; e < NE; ++e) acc[e] += xv * (float)g[e];
  }
#pragma unroll
  for (int e = 0; e < NE; ++e) {
    acc[e] += __shfl_xor(acc[e], 16);
    acc[e] += __shfl_xor(acc[e], 8);
    acc[e] += __shfl_xor(acc[e], 4);
    acc[e] += __shfl_xor(acc[e], 2);
    acc[e] += __shfl_xor(acc[e], 1);
  }
  if (lane == 0) {
    float m = acc[0];
#pragma unroll
    for (int e = 1; e < NE; ++e) m = fmaxf(m, acc[e]);
    float p[NE], sum = 0.f;
#pragma unroll
    for (int e = 0; e < NE; ++e) { p[e] = __expf(acc[e] - m); sum += p[e]; }
    float inv = 1.f / sum;
    int e0 = 0; float b0 = -1.f;
#pragma unroll
    for (int e = 0; e < NE; ++e) if (p[e] > b0) { b0 = p[e]; e0 = e; }
    int e1 = (e0 == 0) ? 1 : 0; float b1 = -1.f;
#pragma unroll
    for (int e = 0; e < NE; ++e) if (e != e0 && p[e] > b1) { b1 = p[e]; e1 = e; }
    eid[s * 2 + 0] = e0; prob[s * 2 + 0] = b0 * inv;
    eid[s * 2 + 1] = e1; prob[s * 2 + 1] = b1 * inv;
    atomicAdd(&cnt[e0], 1);
    atomicAdd(&cnt[e1], 1);
  }
}

__global__ void scan_k(const int* __restrict__ cnt, int* __restrict__ eoff) {
  if (threadIdx.x == 0 && blockIdx.x == 0) {
    int acc = 0;
    for (int e = 0; e < NE; ++e) { eoff[e] = acc; acc += cnt[e]; }
    eoff[NE] = acc;
  }
}

__global__ void scatter_k(const int* __restrict__ eid, const float* __restrict__ prob,
                          const int* __restrict__ eoff, int* __restrict__ cnt2,
                          int* __restrict__ rtok, float* __restrict__ rprob) {
  int s = blockIdx.x * blockDim.x + threadIdx.x;
  if (s >= S_TOK) return;
#pragma unroll
  for (int k = 0; k < TOPK; ++k) {
    int e = eid[s * TOPK + k];
    int pos = eoff[e] + atomicAdd(&cnt2[e], 1);
    rtok[pos] = s;
    rprob[pos] = prob[s * TOPK + k];
  }
}

// ---------------- pass 1: hidden = silu(x@Wg^T) * (x@Wu^T), per expert ----------------
__global__ __launch_bounds__(256) void pass1_k(const __bf16* __restrict__ x,
                                               const __bf16* __restrict__ wg,
                                               const __bf16* __restrict__ wu,
                                               const int* __restrict__ rtok,
                                               const int* __restrict__ eoff,
                                               __bf16* __restrict__ hidden) {
  __shared__ __attribute__((aligned(16))) char smem[51200];
  __bf16* Xs  = (__bf16*)(smem);            // 2 bufs * 64*KST  = 10240 B
  __bf16* WgS = (__bf16*)(smem + 10240);    // 2 bufs * 128*KST = 20480 B
  __bf16* WuS = (__bf16*)(smem + 30720);    // 20480 B
  __bf16* Hs  = (__bf16*)(smem + 10240);    // epilogue staging 64x136 bf16 (overlaps WgS)

  const int e = blockIdx.z;
  const int base = eoff[e];
  const int cnt = eoff[e + 1] - base;
  const int m0 = blockIdx.x * BM;
  if (m0 >= cnt) return;                    // uniform early-exit
  const int h0 = blockIdx.y * BN;

  const int tid = threadIdx.x;
  const int lane = tid & 31;
  const int wv = tid >> 5;
  const int mw = wv & 1;                    // row-group of this wave (2 x 32 rows)
  const int nw = wv >> 1;                   // col-group of this wave (4 x 32 cols)

  // gathered X async-load addressing: thread -> (row 0..63, 16B chunk 0..3)
  const int xrow = tid >> 2, xchk = tid & 3;
  int rr = m0 + xrow; if (rr >= cnt) rr = cnt - 1;
  const uint64_t xga = (uint64_t)(uintptr_t)(x + (size_t)rtok[base + rr] * DDIM + xchk * 8);
  const uint32_t xld = (uint32_t)(uintptr_t)(Xs + xrow * KST + xchk * 8);

  // weight async-load addressing: 128 rows * 4 chunks = 2 per thread per matrix
  const int wr = tid >> 2, wc = tid & 3;
  const size_t wbase = (size_t)e * HDIM;
  const uint64_t gga0 = (uint64_t)(uintptr_t)(wg + (wbase + h0 + wr) * DDIM + wc * 8);
  const uint64_t gga1 = (uint64_t)(uintptr_t)(wg + (wbase + h0 + wr + 64) * DDIM + wc * 8);
  const uint64_t uga0 = (uint64_t)(uintptr_t)(wu + (wbase + h0 + wr) * DDIM + wc * 8);
  const uint64_t uga1 = (uint64_t)(uintptr_t)(wu + (wbase + h0 + wr + 64) * DDIM + wc * 8);
  const uint32_t gld0 = (uint32_t)(uintptr_t)(WgS + wr * KST + wc * 8);
  const uint32_t gld1 = (uint32_t)(uintptr_t)(WgS + (wr + 64) * KST + wc * 8);
  const uint32_t uld0 = (uint32_t)(uintptr_t)(WuS + wr * KST + wc * 8);
  const uint32_t uld1 = (uint32_t)(uintptr_t)(WuS + (wr + 64) * KST + wc * 8);

  const v8f vz = {0.f, 0.f, 0.f, 0.f, 0.f, 0.f, 0.f, 0.f};
  v8f accG[2][2], accU[2][2];
#pragma unroll
  for (int i = 0; i < 2; ++i)
#pragma unroll
    for (int j = 0; j < 2; ++j) { accG[i][j] = vz; accU[i][j] = vz; }

  // prefetch issue for stage kn into buffer b (5 async ops / thread / stage)
  auto issue = [&](int kn, uint32_t b) {
    const uint64_t ko = (uint64_t)kn * (BK * 2);
    async_b128(xld + b * XBUFB, xga + ko);
    async_b128(gld0 + b * WBUFB, gga0 + ko);
    async_b128(gld1 + b * WBUFB, gga1 + ko);
    async_b128(uld0 + b * WBUFB, uga0 + ko);
    async_b128(uld1 + b * WBUFB, uga1 + ko);
  };
  // compute one staged k-slab: ALL fragment loads first (sched-fenced), then 8 WMMAs
  auto compute = [&](const __bf16* Xb, const __bf16* Gb, const __bf16* Ub) {
    v16bf a0 = afrag(Xb + (mw * 32) * KST, lane);
    v16bf a1 = afrag(Xb + (mw * 32 + 16) * KST, lane);
    v16bf bg0 = bfrag(Gb + (nw * 32) * KST, lane);
    v16bf bg1 = bfrag(Gb + (nw * 32 + 16) * KST, lane);
    v16bf bu0 = bfrag(Ub + (nw * 32) * KST, lane);
    v16bf bu1 = bfrag(Ub + (nw * 32 + 16) * KST, lane);
    __builtin_amdgcn_sched_barrier(0);      // group: 12 ds_loads | 8 wmmas
    accG[0][0] = wmma_bf16(a0, bg0, accG[0][0]);
    accG[1][0] = wmma_bf16(a1, bg0, accG[1][0]);
    accU[0][0] = wmma_bf16(a0, bu0, accU[0][0]);
    accU[1][0] = wmma_bf16(a1, bu0, accU[1][0]);
    accG[0][1] = wmma_bf16(a0, bg1, accG[0][1]);
    accG[1][1] = wmma_bf16(a1, bg1, accG[1][1]);
    accU[0][1] = wmma_bf16(a0, bu1, accU[0][1]);
    accU[1][1] = wmma_bf16(a1, bu1, accU[1][1]);
  };

  issue(0, 0);                              // prologue into buffer 0
  const int KS = DDIM / BK;                 // 64 k-steps (even)
  for (int k = 0; k < KS; k += 2) {
    // ---- even step: compute buffer 0, prefetch k+1 into buffer 1 ----
    issue(k + 1, 1);
    wait_async<5>();
    __syncthreads();
    compute(Xs, WgS, WuS);
    __syncthreads();
    // ---- odd step: compute buffer 1, prefetch k+2 into buffer 0 ----
    if (k + 2 < KS) { issue(k + 2, 0); wait_async<5>(); }
    else            { wait_async<0>(); }
    __syncthreads();
    compute(Xs + BM * KST, WgS + BN * KST, WuS + BN * KST);
    __syncthreads();
  }

  // epilogue: silu(gate)*up -> bf16, stage in LDS (overlaps WgS), coalesced store
#pragma unroll
  for (int ms = 0; ms < 2; ++ms)
#pragma unroll
    for (int ns = 0; ns < 2; ++ns) {
      v8f g = accG[ms][ns], u = accU[ms][ns];
#pragma unroll
      for (int v = 0; v < 8; ++v) {
        float gv = g[v];
        float hv = gv / (1.f + __expf(-gv)) * u[v];
        int M = mw * 32 + ms * 16 + ((lane >> 4) << 3) + v;
        int N = nw * 32 + ns * 16 + (lane & 15);
        Hs[M * 136 + N] = (__bf16)hv;
      }
    }
  __syncthreads();
  for (int i = tid; i < BM * (BN / 8); i += 256) {
    int row = i >> 4, c = i & 15;
    if (m0 + row < cnt)
      *(bf8_t*)(hidden + (size_t)(base + m0 + row) * HDIM + h0 + c * 8) =
          *(const bf8_t*)(Hs + row * 136 + c * 8);
  }
}

// ---------------- pass 2: out += prob * (hidden @ Wd^T), per expert ----------------
__global__ __launch_bounds__(256) void pass2_k(const __bf16* __restrict__ hidden,
                                               const __bf16* __restrict__ wd,
                                               const int* __restrict__ rtok,
                                               const float* __restrict__ rprob,
                                               const int* __restrict__ eoff,
                                               float* __restrict__ outf) {
  __shared__ __attribute__((aligned(16))) char smem[64512];
  __bf16* Hb  = (__bf16*)(smem);            // 2 * 64*KST  = 10240 B
  __bf16* WdS = (__bf16*)(smem + 10240);    // 2 * 128*KST = 20480 B
  float*  Os  = (float*)(smem + 30720);     // 64 x 132 f32 = 33792 B

  const int e = blockIdx.z;
  const int base = eoff[e];
  const int cnt = eoff[e + 1] - base;
  const int m0 = blockIdx.x * BM;
  if (m0 >= cnt) return;
  const int d0 = blockIdx.y * BN;

  const int tid = threadIdx.x;
  const int lane = tid & 31;
  const int wv = tid >> 5;
  const int mw = wv & 1, nw = wv >> 1;

  // hidden rows are contiguous by position
  const int xrow = tid >> 2, xchk = tid & 3;
  int rr = m0 + xrow; if (rr >= cnt) rr = cnt - 1;
  const uint64_t aga = (uint64_t)(uintptr_t)(hidden + (size_t)(base + rr) * HDIM + xchk * 8);
  const uint32_t ald = (uint32_t)(uintptr_t)(Hb + xrow * KST + xchk * 8);

  const int wr = tid >> 2, wc = tid & 3;
  const size_t wbase = (size_t)e * DDIM;
  const uint64_t dga0 = (uint64_t)(uintptr_t)(wd + (wbase + d0 + wr) * HDIM + wc * 8);
  const uint64_t dga1 = (uint64_t)(uintptr_t)(wd + (wbase + d0 + wr + 64) * HDIM + wc * 8);
  const uint32_t dld0 = (uint32_t)(uintptr_t)(WdS + wr * KST + wc * 8);
  const uint32_t dld1 = (uint32_t)(uintptr_t)(WdS + (wr + 64) * KST + wc * 8);

  const v8f vz = {0.f, 0.f, 0.f, 0.f, 0.f, 0.f, 0.f, 0.f};
  v8f acc[2][2];
#pragma unroll
  for (int i = 0; i < 2; ++i)
#pragma unroll
    for (int j = 0; j < 2; ++j) acc[i][j] = vz;

  auto issue = [&](int kn, uint32_t b) {
    const uint64_t ko = (uint64_t)kn * (BK * 2);
    async_b128(ald + b * XBUFB, aga + ko);
    async_b128(dld0 + b * WBUFB, dga0 + ko);
    async_b128(dld1 + b * WBUFB, dga1 + ko);
  };
  auto compute = [&](const __bf16* Ab, const __bf16* Bb) {
    v16bf a0 = afrag(Ab + (mw * 32) * KST, lane);
    v16bf a1 = afrag(Ab + (mw * 32 + 16) * KST, lane);
    v16bf bd0 = bfrag(Bb + (nw * 32) * KST, lane);
    v16bf bd1 = bfrag(Bb + (nw * 32 + 16) * KST, lane);
    __builtin_amdgcn_sched_barrier(0);      // group: 8 ds_loads | 4 wmmas
    acc[0][0] = wmma_bf16(a0, bd0, acc[0][0]);
    acc[1][0] = wmma_bf16(a1, bd0, acc[1][0]);
    acc[0][1] = wmma_bf16(a0, bd1, acc[0][1]);
    acc[1][1] = wmma_bf16(a1, bd1, acc[1][1]);
  };

  issue(0, 0);
  const int KS = HDIM / BK;                 // 176 k-steps (even)
  for (int k = 0; k < KS; k += 2) {
    issue(k + 1, 1);
    wait_async<3>();
    __syncthreads();
    compute(Hb, WdS);
    __syncthreads();
    if (k + 2 < KS) { issue(k + 2, 0); wait_async<3>(); }
    else            { wait_async<0>(); }
    __syncthreads();
    compute(Hb + BM * KST, WdS + BN * KST);
    __syncthreads();
  }

  // epilogue: stage f32 tile, then prob-scaled atomic accumulation into outf
#pragma unroll
  for (int ms = 0; ms < 2; ++ms)
#pragma unroll
    for (int ns = 0; ns < 2; ++ns) {
      v8f a = acc[ms][ns];
#pragma unroll
      for (int v = 0; v < 8; ++v) {
        int M = mw * 32 + ms * 16 + ((lane >> 4) << 3) + v;
        int N = nw * 32 + ns * 16 + (lane & 15);
        Os[M * 132 + N] = a[v];
      }
    }
  __syncthreads();
  for (int i = tid; i < BM * BN; i += 256) {
    int row = i >> 7, col = i & 127;
    int m = m0 + row;
    if (m < cnt) {
      float v = Os[row * 132 + col] * rprob[base + m];
      atomicAdd(outf + (size_t)rtok[base + m] * DDIM + d0 + col, v);
    }
  }
}

// ---------------- host launcher ----------------
extern "C" void kernel_launch(void* const* d_in, const int* in_sizes, int n_in,
                              void* d_out, int out_size, void* d_ws, size_t ws_size,
                              hipStream_t stream) {
  (void)in_sizes; (void)n_in; (void)out_size; (void)ws_size;
  const __bf16* x  = (const __bf16*)d_in[0];   // [S, D]
  const __bf16* gk = (const __bf16*)d_in[1];   // [D, E]
  const __bf16* wu = (const __bf16*)d_in[2];   // up_proj   [E, H, D]
  const __bf16* wg = (const __bf16*)d_in[3];   // gate_proj [E, H, D]
  const __bf16* wd = (const __bf16*)d_in[4];   // down_proj [E, D, H]

  char* ws = (char*)d_ws;
  const size_t OUTF = (size_t)S_TOK * DDIM * 4;      // fp32 accumulation buffer
  float* outf  = (float*)ws;
  int*   cnt   = (int*)(ws + OUTF);
  int*   cnt2  = (int*)(ws + OUTF + 256);
  int*   eoff  = (int*)(ws + OUTF + 512);
  int*   eid   = (int*)(ws + OUTF + 1024);
  float* prob  = (float*)(ws + OUTF + 1024 + (size_t)S_TOK * TOPK * 4);
  int*   rtok  = (int*)(ws + OUTF + 1024 + (size_t)S_TOK * TOPK * 8);
  float* rprob = (float*)(ws + OUTF + 1024 + (size_t)S_TOK * TOPK * 12);
  __bf16* hidden = (__bf16*)(ws + OUTF + 1024 + (size_t)S_TOK * TOPK * 16);

  // zero fp32 accumulator + all counters (covers [0, OUTF+1024))
  zero_k<<<2048, 256, 0, stream>>>((uint4*)ws, (long long)((OUTF + 1024) / 16));
  router_k<<<S_TOK / 8, 256, 0, stream>>>(x, gk, eid, prob, cnt);
  scan_k<<<1, 32, 0, stream>>>(cnt, eoff);
  scatter_k<<<S_TOK / 256, 256, 0, stream>>>(eid, prob, eoff, cnt2, rtok, rprob);
  pass1_k<<<dim3(S_TOK / BM, HDIM / BN, NE), 256, 0, stream>>>(x, wg, wu, rtok, eoff, hidden);
  pass2_k<<<dim3(S_TOK / BM, DDIM / BN, NE), 256, 0, stream>>>(hidden, wd, rtok, rprob, eoff, outf);
  finalize_k<<<2048, 256, 0, stream>>>(outf, (__bf16*)d_out, (long long)S_TOK * DDIM);
}
